// Model_63651415327152
// MI455X (gfx1250) — matrix operand order repaired and checked by reading, not compiled, not run on hardware
//
#include <hip/hip_runtime.h>
#include <math.h>

typedef __attribute__((ext_vector_type(16))) __bf16 v16bf;
typedef __attribute__((ext_vector_type(8)))  float  v8f;

#define B_   2
#define H_   16
#define SQ_  2048
#define SK_  2048
#define D_   128

#define BQ   128  // query rows per workgroup (8 waves x 16 rows)
#define BK   64   // key rows per tile
#define NW   8    // waves per workgroup (wave32)
#define NT   (SK_ / BK)

// softmax scale 1/0.05 = 20, folded with log2(e) so we can use exp2 (v_exp_f32)
static constexpr float QSCALE = 20.0f * 1.4426950408889634f;

struct __attribute__((aligned(16))) U128 { unsigned int w[4]; };
struct U256 { U128 lo, hi; };

// A-matrix fragment (16x32 bf16): lane-half hi=0 -> K[0..7],[16..23]; hi=1 -> K[8..15],[24..31]
__device__ inline v16bf load_frag_a(const __bf16* rowp, int hi) {
    const __bf16* p = rowp + hi * 8;
    U256 t;
    t.lo = *reinterpret_cast<const U128*>(p);
    t.hi = *reinterpret_cast<const U128*>(p + 16);
    return __builtin_bit_cast(v16bf, t);
}

// B-matrix fragment (32x16 bf16): lane-half hi=0 -> K[0..7],[16..23]; hi=1 -> K[8..15],[24..31]
__device__ inline v16bf load_frag_b(const __bf16* rowp, int hi) {
    const __bf16* p = rowp + hi * 8;
    U256 t;
    t.lo = *reinterpret_cast<const U128*>(p);
    t.hi = *reinterpret_cast<const U128*>(p + 16);
    return __builtin_bit_cast(v16bf, t);
}

__global__ __launch_bounds__(256) void fa_fwd_kernel(
    const float* __restrict__ q, const float* __restrict__ k,
    const float* __restrict__ v, const unsigned char* __restrict__ pmask,
    float* __restrict__ out)
{
    __shared__ alignas(32) __bf16 Ks[BK][D_];      // 16 KB
    __shared__ alignas(32) __bf16 Vt[D_][BK];      // 16 KB (transposed V)
    __shared__ alignas(32) __bf16 Ps[NW][16][BK];  // 16 KB per-wave P scratch

    const int tid  = threadIdx.x;
    const int wave = tid >> 5;
    const int lane = tid & 31;
    const int l16  = lane & 15;
    const int hi   = lane >> 4;

    const int bh    = blockIdx.y;
    const int qtile = blockIdx.x;
    const size_t head_base = (size_t)bh * SQ_ * D_;   // SQ_ == SK_
    const float* kg = k + head_base;
    const float* vg = v + head_base;

    // ---- hoist Q A-fragments straight from global (done once; pre-scaled) ----
    v16bf qf[4];
    {
        const float* qrow = q + head_base
                          + (size_t)(qtile * BQ + wave * 16 + l16) * D_;
        #pragma unroll
        for (int c = 0; c < 4; ++c) {
            const float* p = qrow + c * 32 + hi * 8;
            float4 a  = *reinterpret_cast<const float4*>(p);
            float4 b  = *reinterpret_cast<const float4*>(p + 4);
            float4 c2 = *reinterpret_cast<const float4*>(p + 16);
            float4 d2 = *reinterpret_cast<const float4*>(p + 20);
            v16bf r;
            r[0]  = (__bf16)(a.x  * QSCALE); r[1]  = (__bf16)(a.y  * QSCALE);
            r[2]  = (__bf16)(a.z  * QSCALE); r[3]  = (__bf16)(a.w  * QSCALE);
            r[4]  = (__bf16)(b.x  * QSCALE); r[5]  = (__bf16)(b.y  * QSCALE);
            r[6]  = (__bf16)(b.z  * QSCALE); r[7]  = (__bf16)(b.w  * QSCALE);
            r[8]  = (__bf16)(c2.x * QSCALE); r[9]  = (__bf16)(c2.y * QSCALE);
            r[10] = (__bf16)(c2.z * QSCALE); r[11] = (__bf16)(c2.w * QSCALE);
            r[12] = (__bf16)(d2.x * QSCALE); r[13] = (__bf16)(d2.y * QSCALE);
            r[14] = (__bf16)(d2.z * QSCALE); r[15] = (__bf16)(d2.w * QSCALE);
            qf[c] = r;
        }
    }

    // online-softmax state: this lane tracks rows (r + 8*hi) of its 16-row tile
    float m_run[8], l_run[8];
    #pragma unroll
    for (int r = 0; r < 8; ++r) { m_run[r] = -__builtin_inff(); l_run[r] = 0.0f; }
    v8f oacc[8];
    #pragma unroll
    for (int dt = 0; dt < 8; ++dt) oacc[dt] = (v8f){0,0,0,0,0,0,0,0};

    // ---- register-staged pipeline: issue tile-0 global loads now ----
    float4 kst[8], vst[8];
    #pragma unroll
    for (int i = 0; i < 8; ++i) {
        int idx = i * 256 + tid;                 // float4 index in 64x128 tile
        kst[i] = reinterpret_cast<const float4*>(kg)[idx];
        vst[i] = reinterpret_cast<const float4*>(vg)[idx];
    }

    for (int kt = 0; kt < NT; ++kt) {
        __syncthreads();   // previous iteration finished reading Ks/Vt
        // ---- drain staged registers into LDS (bf16; V transposed) ----
        #pragma unroll
        for (int i = 0; i < 8; ++i) {
            int idx = i * 256 + tid;
            int row = idx >> 5;                  // 32 float4 per row of 128
            int c4  = idx & 31;
            float4 f = kst[i];
            __bf16* dst = &Ks[row][c4 * 4];
            dst[0] = (__bf16)f.x; dst[1] = (__bf16)f.y;
            dst[2] = (__bf16)f.z; dst[3] = (__bf16)f.w;
            float4 g = vst[i];
            int d0 = c4 * 4;
            Vt[d0 + 0][row] = (__bf16)g.x;
            Vt[d0 + 1][row] = (__bf16)g.y;
            Vt[d0 + 2][row] = (__bf16)g.z;
            Vt[d0 + 3][row] = (__bf16)g.w;
        }
        // ---- issue next tile's global loads (hide under this tile's compute) ----
        if (kt + 1 < NT) {
            const float* kn = kg + (size_t)(kt + 1) * BK * D_;
            const float* vn = vg + (size_t)(kt + 1) * BK * D_;
            #pragma unroll
            for (int i = 0; i < 8; ++i) {
                int idx = i * 256 + tid;
                kst[i] = reinterpret_cast<const float4*>(kn)[idx];
                vst[i] = reinterpret_cast<const float4*>(vn)[idx];
            }
        }
        // ---- prefetch tile kt+2 into L2 (one 128B line per thread per array) ----
        if (kt + 2 < NT) {
            const float* kp = kg + (size_t)(kt + 2) * BK * D_;
            const float* vp = vg + (size_t)(kt + 2) * BK * D_;
            __builtin_prefetch(kp + (size_t)tid * 32, 0, 1);
            __builtin_prefetch(vp + (size_t)tid * 32, 0, 1);
        }
        __syncthreads();

        // ---- S = (scaled Q) K^T : 4 key sub-tiles of 16, K-dim = 128 = 4x32 ----
        v8f s[4];
        #pragma unroll
        for (int nt = 0; nt < 4; ++nt) {
            v8f acc = (v8f){0,0,0,0,0,0,0,0};
            #pragma unroll
            for (int c = 0; c < 4; ++c) {
                v16bf bf = load_frag_b(&Ks[nt * 16 + l16][c * 32], hi);
                acc = __builtin_amdgcn_wmma_f32_16x16x32_bf16(
                          false, qf[c], false, bf, (short)0, acc, false, false);
            }
            s[nt] = acc;
        }

        // ---- online softmax (base-2 domain; scale folded into Q) ----
        float mnew[8], alpha[8], lsum[8];
        #pragma unroll
        for (int r = 0; r < 8; ++r) {
            float mx = fmaxf(fmaxf(s[0][r], s[1][r]), fmaxf(s[2][r], s[3][r]));
            for (int off = 1; off < 16; off <<= 1)
                mx = fmaxf(mx, __shfl_xor(mx, off));
            mnew[r]  = fmaxf(m_run[r], mx);
            alpha[r] = exp2f(m_run[r] - mnew[r]);   // first iter: exp2(-inf)=0
            lsum[r]  = 0.0f;
        }
        #pragma unroll
        for (int nt = 0; nt < 4; ++nt) {
            #pragma unroll
            for (int r = 0; r < 8; ++r) {
                float p = exp2f(s[nt][r] - mnew[r]);
                lsum[r] += p;
                Ps[wave][r + hi * 8][nt * 16 + l16] = (__bf16)p;
            }
        }
        #pragma unroll
        for (int r = 0; r < 8; ++r) {
            float ls = lsum[r];
            for (int off = 1; off < 16; off <<= 1)
                ls += __shfl_xor(ls, off);
            l_run[r] = l_run[r] * alpha[r] + ls;
            m_run[r] = mnew[r];
        }
        #pragma unroll
        for (int dt = 0; dt < 8; ++dt)
            #pragma unroll
            for (int r = 0; r < 8; ++r)
                oacc[dt][r] *= alpha[r];

        // P scratch is wave-private; drain DS before re-reading in A-layout
        asm volatile("s_wait_dscnt 0" ::: "memory");

        // ---- O += P V : K-dim 64 keys = 2x32, 8 output d-tiles of 16 ----
        #pragma unroll
        for (int kc = 0; kc < 2; ++kc) {
            v16bf pf = load_frag_a(&Ps[wave][l16][kc * 32], hi);
            #pragma unroll
            for (int dt = 0; dt < 8; ++dt) {
                v16bf vf = load_frag_b(&Vt[dt * 16 + l16][kc * 32], hi);
                oacc[dt] = __builtin_amdgcn_wmma_f32_16x16x32_bf16(
                               false, pf, false, vf, (short)0, oacc[dt], false, false);
            }
        }
    }

    // ---- epilogue: normalize, apply padding mask over D, store fp32 ----
    float inv_l[8];
    #pragma unroll
    for (int r = 0; r < 8; ++r) inv_l[r] = 1.0f / l_run[r];

    #pragma unroll
    for (int dt = 0; dt < 8; ++dt) {
        int d = dt * 16 + l16;
        float keep = pmask[d] ? 0.0f : 1.0f;
        #pragma unroll
        for (int r = 0; r < 8; ++r) {
            int row = qtile * BQ + wave * 16 + hi * 8 + r;
            out[head_base + (size_t)row * D_ + d] = oacc[dt][r] * inv_l[r] * keep;
        }
    }
}

extern "C" void kernel_launch(void* const* d_in, const int* in_sizes, int n_in,
                              void* d_out, int out_size, void* d_ws, size_t ws_size,
                              hipStream_t stream) {
    (void)in_sizes; (void)n_in; (void)out_size; (void)d_ws; (void)ws_size;
    const float* q = (const float*)d_in[0];
    const float* k = (const float*)d_in[1];
    const float* v = (const float*)d_in[2];
    const unsigned char* pmask = (const unsigned char*)d_in[3];   // jnp.bool_ -> 1 byte
    float* out = (float*)d_out;

    dim3 grid(SQ_ / BQ, B_ * H_);   // (16, 32)
    fa_fwd_kernel<<<grid, 256, 0, stream>>>(q, k, v, pmask, out);
}
